// ImageModel_35003983463034
// MI455X (gfx1250) — matrix operand order repaired and checked by reading, not compiled, not run on hardware
//
#include <hip/hip_runtime.h>

// ---------------------------------------------------------------------------
// UR-LSTM on MI455X (gfx1250): bf16 WMMA GEMMs + fp32 gate math.
// S=128, B=512, E=128, H=256, R=1024, NCLS=10
// GEMM: 2x4 register tile per wave (8 wmma / 32-K step), double-buffered frags.
// ---------------------------------------------------------------------------

typedef __attribute__((ext_vector_type(16))) __bf16 v16bf;
typedef __attribute__((ext_vector_type(8)))  __bf16 v8bf;
typedef __attribute__((ext_vector_type(8)))  float  v8f;

#define S_    128
#define B_    512
#define E_    128
#define H_    256
#define R_    1024
#define NCLS_ 10
#define G4H   (4 * H_)
#define SH_   (S_ * H_)
#define FB_   1.0f

// float -> bf16, round-to-nearest-even
__device__ __forceinline__ unsigned short f2bf(float f) {
  unsigned int u = __float_as_uint(f);
  u += 0x7FFFu + ((u >> 16) & 1u);
  return (unsigned short)(u >> 16);
}

__global__ void k_f32_to_bf16(const float* __restrict__ src,
                              unsigned short* __restrict__ dst, long long n) {
  long long i  = (long long)blockIdx.x * blockDim.x + threadIdx.x;
  long long st = (long long)gridDim.x * blockDim.x;
  for (; i < n; i += st) dst[i] = f2bf(src[i]);
}

__global__ void k_zero_f32(float* __restrict__ p, long long n) {
  long long i  = (long long)blockIdx.x * blockDim.x + threadIdx.x;
  long long st = (long long)gridDim.x * blockDim.x;
  for (; i < n; i += st) p[i] = 0.0f;
}

__global__ void k_zero_u16(unsigned short* __restrict__ p, long long n) {
  long long i  = (long long)blockIdx.x * blockDim.x + threadIdx.x;
  long long st = (long long)gridDim.x * blockDim.x;
  for (; i < n; i += st) p[i] = 0;
}

// A fragment (16x32 bf16, MxK): lane l<16 -> m=l, K runs {k0..k0+7, k0+16..k0+23};
// lanes 16-31 -> same rows, runs shifted by 8. Two 16B loads per lane.
__device__ __forceinline__ v16bf load_afrag(const unsigned short* __restrict__ p) {
  v8bf lo = *reinterpret_cast<const v8bf*>(p);
  v8bf hi = *reinterpret_cast<const v8bf*>(p + 16);
  v16bf a;
#pragma unroll
  for (int i = 0; i < 8; ++i) { a[i] = lo[i]; a[i + 8] = hi[i]; }
  return a;
}

// ---------------------------------------------------------------------------
// C[M,N] = A[M,K] * Bm[N,K]^T  (+ Cinit[M,N]) (+ bias[N]) (relu?)
// A, Bm bf16 row-major (K contiguous). C fp32.
// Block = 256 threads = 8 waves (4 M x 2 N); wave tile = 32x64 (8 accumulators).
// Grid: (M/128, N/128). Requires M%128==0, N%128==0, K%32==0,
// lda/ldb multiples of 16 elements, 32B-aligned bases.
// ---------------------------------------------------------------------------
__global__ __launch_bounds__(256) void k_gemm_bf16(
    const unsigned short* __restrict__ A, int lda,
    const unsigned short* __restrict__ Bm, int ldb,
    const float* __restrict__ Cinit,
    const float* __restrict__ bias,
    float* __restrict__ C, int ldc,
    int K, int relu)
{
  const int lane = threadIdx.x & 31;
  const int wave = threadIdx.x >> 5;
  const int wm   = wave & 3;   // M sub-block 0..3
  const int wn   = wave >> 2;  // N sub-block 0..1
  const int half = lane >> 4;
  const int l16  = lane & 15;

  const long long m0 = (long long)blockIdx.x * 128 + wm * 32;
  const long long n0 = (long long)blockIdx.y * 128 + wn * 64;

  const v8f vzero = {0.f, 0.f, 0.f, 0.f, 0.f, 0.f, 0.f, 0.f};
  v8f acc[8];
#pragma unroll
  for (int j = 0; j < 8; ++j) acc[j] = vzero;

  // Lane base pointers.
  const unsigned short* ap0 = A + (m0 + l16) * (long long)lda + half * 16;
  const unsigned short* ap1 = ap0 + 16 * (long long)lda;
  const unsigned short* bp0 = Bm + (n0 + 0 * 16 + l16) * (long long)ldb + half * 16;
  const unsigned short* bp1 = bp0 + 16 * (long long)ldb;
  const unsigned short* bp2 = bp0 + 32 * (long long)ldb;
  const unsigned short* bp3 = bp0 + 48 * (long long)ldb;

  // Prime the pipeline with k-tile 0.
  v16bf a0 = *reinterpret_cast<const v16bf*>(ap0);
  v16bf a1 = *reinterpret_cast<const v16bf*>(ap1);
  v16bf b0 = *reinterpret_cast<const v16bf*>(bp0);
  v16bf b1 = *reinterpret_cast<const v16bf*>(bp1);
  v16bf b2 = *reinterpret_cast<const v16bf*>(bp2);
  v16bf b3 = *reinterpret_cast<const v16bf*>(bp3);

  for (int k0 = 0; k0 < K; k0 += 32) {
    // Issue next k-tile's loads before consuming current fragments
    // (clamped offset on the last iteration -> loads stay unconditional).
    const int kn = (k0 + 32 < K) ? (k0 + 32) : 0;
    if (k0 + 256 < K) {  // uniform scalar branch
      __builtin_prefetch((const void*)(ap0 + k0 + 256), 0, 3);
      __builtin_prefetch((const void*)(bp0 + k0 + 256), 0, 3);
    }
    v16bf na0 = *reinterpret_cast<const v16bf*>(ap0 + kn);
    v16bf na1 = *reinterpret_cast<const v16bf*>(ap1 + kn);
    v16bf nb0 = *reinterpret_cast<const v16bf*>(bp0 + kn);
    v16bf nb1 = *reinterpret_cast<const v16bf*>(bp1 + kn);
    v16bf nb2 = *reinterpret_cast<const v16bf*>(bp2 + kn);
    v16bf nb3 = *reinterpret_cast<const v16bf*>(bp3 + kn);

    acc[0] = __builtin_amdgcn_wmma_f32_16x16x32_bf16(false, a0, false, b0, (short)0, acc[0], false, false);
    acc[1] = __builtin_amdgcn_wmma_f32_16x16x32_bf16(false, a0, false, b1, (short)0, acc[1], false, false);
    acc[2] = __builtin_amdgcn_wmma_f32_16x16x32_bf16(false, a0, false, b2, (short)0, acc[2], false, false);
    acc[3] = __builtin_amdgcn_wmma_f32_16x16x32_bf16(false, a0, false, b3, (short)0, acc[3], false, false);
    acc[4] = __builtin_amdgcn_wmma_f32_16x16x32_bf16(false, a1, false, b0, (short)0, acc[4], false, false);
    acc[5] = __builtin_amdgcn_wmma_f32_16x16x32_bf16(false, a1, false, b1, (short)0, acc[5], false, false);
    acc[6] = __builtin_amdgcn_wmma_f32_16x16x32_bf16(false, a1, false, b2, (short)0, acc[6], false, false);
    acc[7] = __builtin_amdgcn_wmma_f32_16x16x32_bf16(false, a1, false, b3, (short)0, acc[7], false, false);

    a0 = na0; a1 = na1;
    b0 = nb0; b1 = nb1; b2 = nb2; b3 = nb3;
  }

  // Epilogue: element i of lane -> (m0 + 16*mi + 8*half + i, n0 + 16*j + l16)
#pragma unroll
  for (int mi = 0; mi < 2; ++mi) {
#pragma unroll
    for (int i = 0; i < 8; ++i) {
      const long long m = m0 + mi * 16 + half * 8 + i;
      float* crow = C + m * (long long)ldc;
#pragma unroll
      for (int j = 0; j < 4; ++j) {
        const long long n = n0 + j * 16 + l16;
        float v = acc[mi * 4 + j][i];
        if (Cinit) v += Cinit[m * (long long)ldc + n];
        if (bias)  v += bias[n];
        if (relu)  v = fmaxf(v, 0.f);
        crow[n] = v;
      }
    }
  }
}

// ---------------------------------------------------------------------------
// UR-LSTM gate update for step t. gates[B,4H] = xg_t + h U^T (precomputed).
// Writes new cell c and bf16 hidden state into hs[b][t*H + j].
// ---------------------------------------------------------------------------
__global__ __launch_bounds__(256) void k_gate_step(
    const float* __restrict__ gates,
    float* __restrict__ c,
    unsigned short* __restrict__ hsbf,
    int t)
{
  int idx = blockIdx.x * blockDim.x + threadIdx.x;  // 0 .. B*H-1
  int bb  = idx >> 8;        // / H  (H=256)
  int j   = idx & (H_ - 1);  // % H
  const float* g = gates + (long long)bb * G4H;
  float f_ = g[j];
  float r_ = g[H_ + j];
  float u_ = g[2 * H_ + j];
  float o_ = g[3 * H_ + j];
  float f  = 1.f / (1.f + __expf(-(f_ + FB_)));
  float r  = 1.f / (1.f + __expf(-(r_ - FB_)));
  float gg = 2.f * r * f + (1.f - 2.f * r) * f * f;
  float cn = gg * c[idx] + (1.f - gg) * tanhf(u_);
  c[idx] = cn;
  float h = (1.f / (1.f + __expf(-o_))) * tanhf(cn);
  hsbf[(long long)bb * SH_ + (long long)t * H_ + j] = f2bf(h);
}

// out[B,10] = y[B,R] @ W2[10,R]^T + b2  (N=10 < WMMA tile; scalar fp32)
__global__ void k_final(const float* __restrict__ y, const float* __restrict__ W2,
                        const float* __restrict__ b2, float* __restrict__ out) {
  int idx = blockIdx.x * blockDim.x + threadIdx.x;
  if (idx >= B_ * NCLS_) return;
  int bb = idx / NCLS_, n = idx % NCLS_;
  const float* yr = y + (long long)bb * R_;
  const float* wr = W2 + (long long)n * R_;
  float s = b2[n];
  for (int k = 0; k < R_; k += 4)
    s += yr[k] * wr[k] + yr[k + 1] * wr[k + 1] +
         yr[k + 2] * wr[k + 2] + yr[k + 3] * wr[k + 3];
  out[idx] = s;
}

extern "C" void kernel_launch(void* const* d_in, const int* in_sizes, int n_in,
                              void* d_out, int out_size, void* d_ws, size_t ws_size,
                              hipStream_t stream) {
  (void)in_sizes; (void)n_in; (void)out_size; (void)ws_size;
  const float* x  = (const float*)d_in[0];   // [S,B,E]
  const float* W  = (const float*)d_in[1];   // [4H,E]
  const float* U  = (const float*)d_in[2];   // [4H,H]
  const float* bg = (const float*)d_in[3];   // [4H]
  const float* W1 = (const float*)d_in[4];   // [R,S*H]
  const float* b1 = (const float*)d_in[5];   // [R]
  const float* W2 = (const float*)d_in[6];   // [10,R]
  const float* b2 = (const float*)d_in[7];   // [10]
  float* out = (float*)d_out;

  // Workspace carve (256B aligned)
  char* p = (char*)d_ws;
  auto carve = [&](size_t bytes) {
    char* r = p;
    p += (bytes + 255) & ~(size_t)255;
    return (void*)r;
  };
  unsigned short* xbf  = (unsigned short*)carve((size_t)S_ * B_ * E_ * 2);
  unsigned short* Wbf  = (unsigned short*)carve((size_t)G4H * E_ * 2);
  unsigned short* Ubf  = (unsigned short*)carve((size_t)G4H * H_ * 2);
  unsigned short* W1bf = (unsigned short*)carve((size_t)R_ * SH_ * 2);
  unsigned short* hsbf = (unsigned short*)carve((size_t)B_ * SH_ * 2);
  unsigned short* hz   = (unsigned short*)carve((size_t)B_ * H_ * 2);
  float* xg    = (float*)carve((size_t)S_ * B_ * G4H * 4);
  float* cbuf  = (float*)carve((size_t)B_ * H_ * 4);
  float* gates = (float*)carve((size_t)B_ * G4H * 4);
  float* ybuf  = (float*)carve((size_t)B_ * R_ * 4);

  // 1) fp32 -> bf16 conversions
  k_f32_to_bf16<<<2048, 256, 0, stream>>>(x,  xbf,  (long long)S_ * B_ * E_);
  k_f32_to_bf16<<<64,   256, 0, stream>>>(W,  Wbf,  (long long)G4H * E_);
  k_f32_to_bf16<<<128,  256, 0, stream>>>(U,  Ubf,  (long long)G4H * H_);
  k_f32_to_bf16<<<4096, 256, 0, stream>>>(W1, W1bf, (long long)R_ * SH_);
  k_zero_f32<<<128, 256, 0, stream>>>(cbuf, (long long)B_ * H_);
  k_zero_u16<<<128, 256, 0, stream>>>(hz,   (long long)B_ * H_);

  // 2) xg[S*B, 4H] = x @ W^T + b     grid (65536/128, 1024/128)
  k_gemm_bf16<<<dim3((S_ * B_) / 128, G4H / 128), 256, 0, stream>>>(
      xbf, E_, Wbf, E_, /*Cinit=*/nullptr, /*bias=*/bg, xg, G4H, E_, /*relu=*/0);

  // 3) recurrence: 128 sequential (GEMM + gate) steps
  for (int t = 0; t < S_; ++t) {
    const unsigned short* A = (t == 0) ? hz : (hsbf + (size_t)(t - 1) * H_);
    int lda = (t == 0) ? H_ : SH_;
    k_gemm_bf16<<<dim3(B_ / 128, G4H / 128), 256, 0, stream>>>(
        A, lda, Ubf, H_, /*Cinit=*/xg + (size_t)t * B_ * G4H, /*bias=*/nullptr,
        gates, G4H, H_, /*relu=*/0);
    k_gate_step<<<(B_ * H_) / 256, 256, 0, stream>>>(gates, cbuf, hsbf, t);
  }

  // 4) y[B,R] = relu(hs[B,S*H] @ W1^T + b1)
  k_gemm_bf16<<<dim3(B_ / 128, R_ / 128), 256, 0, stream>>>(
      hsbf, SH_, W1bf, SH_, /*Cinit=*/nullptr, /*bias=*/b1, ybuf, R_, SH_, /*relu=*/1);

  // 5) out[B,10] = y @ W2^T + b2
  k_final<<<(B_ * NCLS_ + 255) / 256, 256, 0, stream>>>(ybuf, W2, b2, out);
}
